// attention_as_rnn_31937376813365
// MI455X (gfx1250) — compile-verified, matrix operand order checked by reading, not run on hardware
//
#include <hip/hip_runtime.h>

// ---------------------------------------------------------------------------
// Problem constants (from reference)
// ---------------------------------------------------------------------------
#define BB       4
#define TT       2048
#define HH       16
#define DD       64
#define DIM_IN   1024
#define NKV      2048          // HH * DD * 2
#define DIM_OUT  1024
#define MROWS    (BB * TT)     // 8192
#define NCHUNK   16
#define CLEN     (TT / NCHUNK) // 128

typedef float v8f  __attribute__((ext_vector_type(8)));
typedef __bf16 v16bf __attribute__((ext_vector_type(16)));

struct FragBF16 { union { unsigned int u[8]; v16bf v; }; };

__device__ __forceinline__ unsigned short f32_to_bf16_rne(float f) {
    unsigned int x = __float_as_uint(f);
    unsigned int r = x + 0x7fffu + ((x >> 16) & 1u);
    return (unsigned short)(r >> 16);
}

// Load one 16x32 bf16 fragment (A-layout; identical striping for B^T rows).
// lane<16  : row = lr, K-chunks [k0..k0+7]  and [k0+16..k0+23]
// lane>=16 : row = lr, K-chunks [k0+8..+15] and [k0+24..+31]
__device__ __forceinline__ void load_frag(FragBF16& f, const unsigned short* p) {
    uint4 lo = *(const uint4*)p;
    uint4 hi = *(const uint4*)(p + 16);
    f.u[0] = lo.x; f.u[1] = lo.y; f.u[2] = lo.z; f.u[3] = lo.w;
    f.u[4] = hi.x; f.u[5] = hi.y; f.u[6] = hi.z; f.u[7] = hi.w;
}

// ---------------------------------------------------------------------------
// f32 -> bf16 bulk convert
// ---------------------------------------------------------------------------
__global__ void cvt_bf16_kernel(const float* __restrict__ src,
                                unsigned short* __restrict__ dst, int n) {
    int i = blockIdx.x * blockDim.x + threadIdx.x;
    if (i < n) dst[i] = f32_to_bf16_rne(src[i]);
}

// ---------------------------------------------------------------------------
// kv_kernel (K=1024 x N=2048, row-major) -> bf16 W^T (N x K), LDS tiled
// ---------------------------------------------------------------------------
__global__ void transpose_cvt_kernel(const float* __restrict__ w,
                                     unsigned short* __restrict__ wt) {
    __shared__ unsigned short tile[32][33];
    int nb = blockIdx.x * 32;   // N tile
    int kb = blockIdx.y * 32;   // K tile
    int tx = threadIdx.x, ty = threadIdx.y; // 32 x 8
#pragma unroll
    for (int j = 0; j < 32; j += 8)
        tile[ty + j][tx] = f32_to_bf16_rne(w[(size_t)(kb + ty + j) * NKV + nb + tx]);
    __syncthreads();
#pragma unroll
    for (int j = 0; j < 32; j += 8)
        wt[(size_t)(nb + ty + j) * DIM_IN + kb + tx] = tile[tx][ty + j];
}

// ---------------------------------------------------------------------------
// WMMA bf16 GEMM:  C(MxN,f32) = A(MxK,bf16 row-major) * Bt(NxK,bf16 row-major)^T
// Wave tile 32x32 (2x2 WMMA accs), 8 waves/block arranged 2(M) x 4(N)
// Block tile: 64(M) x 128(N).  Optional bias (length N).
// ---------------------------------------------------------------------------
__global__ __launch_bounds__(256)
void wmma_gemm_kernel(const unsigned short* __restrict__ A,
                      const unsigned short* __restrict__ Bt,
                      float* __restrict__ C,
                      const float* __restrict__ bias,
                      int N, int K) {
    int lane  = threadIdx.x & 31;
    int wave  = threadIdx.x >> 5;
    int waveM = wave & 1, waveN = wave >> 1;
    int m0 = blockIdx.x * 64  + waveM * 32;
    int n0 = blockIdx.y * 128 + waveN * 32;
    int half = lane >> 4;       // 0/1: selects the K-chunk phase
    int lr   = lane & 15;

    v8f acc[2][2] = {};
    for (int k0 = 0; k0 < K; k0 += 32) {
        int kc = k0 + half * 8;
        FragBF16 a[2], b[2];
#pragma unroll
        for (int i = 0; i < 2; ++i) {
            load_frag(a[i], A  + (size_t)(m0 + i * 16 + lr) * K + kc);
            load_frag(b[i], Bt + (size_t)(n0 + i * 16 + lr) * K + kc);
        }
#pragma unroll
        for (int i = 0; i < 2; ++i)
#pragma unroll
            for (int j = 0; j < 2; ++j)
                acc[i][j] = __builtin_amdgcn_wmma_f32_16x16x32_bf16(
                    false, a[i].v, false, b[j].v, (short)0, acc[i][j], false, false);
    }
    // C/D layout: VGPR r, lanes<16 -> row r, lanes>=16 -> row r+8; col = lane&15
#pragma unroll
    for (int i = 0; i < 2; ++i)
#pragma unroll
        for (int j = 0; j < 2; ++j) {
            int col = n0 + j * 16 + lr;
            float bv = bias ? bias[col] : 0.0f;
#pragma unroll
            for (int r = 0; r < 8; ++r) {
                int row = m0 + i * 16 + r + half * 8;
                C[(size_t)row * N + col] = acc[i][j][r] + bv;
            }
        }
}

// ---------------------------------------------------------------------------
// Scan pass 1: per-(b,h,chunk) carry totals.
// One wave per (b,h,chunk); lane owns dims {lane, lane+32}.
// kv layout: row (b*T+t), col (h*128 + d*2 + {k,v})  -> float2 per dim.
// ---------------------------------------------------------------------------
__device__ __forceinline__ float wave_allreduce_add(float x) {
#pragma unroll
    for (int off = 16; off > 0; off >>= 1) x += __shfl_xor(x, off, 32);
    return x;
}

__global__ __launch_bounds__(128)
void scan_carry_kernel(const float* __restrict__ kv,
                       const float* __restrict__ qk,
                       float* __restrict__ carry) {
    int wid   = (blockIdx.x * blockDim.x + threadIdx.x) >> 5;
    int lane  = threadIdx.x & 31;
    int chunk = wid & (NCHUNK - 1);
    int bh    = wid >> 4;
    int b = bh >> 4, h = bh & 15;
    float q0 = qk[h * DD + lane], q1 = qk[h * DD + lane + 32];
    float m = -__builtin_inff(), u = 0.0f, w0 = 0.0f, w1 = 0.0f;
    int t0 = chunk * CLEN;
    for (int i = 0; i < CLEN; ++i) {
        int t = t0 + i;
        const float2* p = (const float2*)(kv + (size_t)(b * TT + t) * NKV + h * 128);
        float2 kv0 = p[lane];
        float2 kv1 = p[lane + 32];
        float dot = wave_allreduce_add(q0 * kv0.x + q1 * kv1.x);
        float mn = fmaxf(m, dot);
        float e0 = __expf(m - mn), e1 = __expf(dot - mn);
        u  = u  * e0 + e1;
        w0 = w0 * e0 + kv0.y * e1;
        w1 = w1 * e0 + kv1.y * e1;
        m = mn;
    }
    float* c = carry + (size_t)(bh * NCHUNK + chunk) * 68;
    if (lane == 0) { c[0] = m; c[1] = u; }
    c[2 + lane] = w0;
    c[2 + lane + 32] = w1;
}

// ---------------------------------------------------------------------------
// Scan pass 2: exclusive prefix over the 16 chunk carries (per bh). 1 wave/bh.
// ---------------------------------------------------------------------------
__global__ __launch_bounds__(32)
void scan_prefix_kernel(const float* __restrict__ carry,
                        float* __restrict__ prefix) {
    int bh = blockIdx.x;
    int lane = threadIdx.x;
    float pm = -__builtin_inff(), pu = 0.0f, pw0 = 0.0f, pw1 = 0.0f;
    for (int c = 0; c < NCHUNK; ++c) {
        float* pp = prefix + (size_t)(bh * NCHUNK + c) * 68;
        if (lane == 0) { pp[0] = pm; pp[1] = pu; }
        pp[2 + lane] = pw0;
        pp[2 + lane + 32] = pw1;
        const float* cc = carry + (size_t)(bh * NCHUNK + c) * 68;
        float cm = cc[0], cu = cc[1];
        float cw0 = cc[2 + lane], cw1 = cc[2 + lane + 32];
        float mn = fmaxf(pm, cm);
        float ep = __expf(pm - mn), ec = __expf(cm - mn);
        pu  = pu  * ep + cu  * ec;
        pw0 = pw0 * ep + cw0 * ec;
        pw1 = pw1 * ep + cw1 * ec;
        pm = mn;
    }
}

// ---------------------------------------------------------------------------
// Scan pass 3: inclusive scan from exclusive prefix; emit h = w/u per head.
// hbuf layout head-major: ((h*B + b)*T + t)*64 + d  (coalesced head reduce)
// ---------------------------------------------------------------------------
__global__ __launch_bounds__(128)
void scan_apply_kernel(const float* __restrict__ kv,
                       const float* __restrict__ qk,
                       const float* __restrict__ prefix,
                       float* __restrict__ hbuf) {
    int wid   = (blockIdx.x * blockDim.x + threadIdx.x) >> 5;
    int lane  = threadIdx.x & 31;
    int chunk = wid & (NCHUNK - 1);
    int bh    = wid >> 4;
    int b = bh >> 4, h = bh & 15;
    float q0 = qk[h * DD + lane], q1 = qk[h * DD + lane + 32];
    const float* pp = prefix + (size_t)(bh * NCHUNK + chunk) * 68;
    float m = pp[0], u = pp[1];
    float w0 = pp[2 + lane], w1 = pp[2 + lane + 32];
    int t0 = chunk * CLEN;
    for (int i = 0; i < CLEN; ++i) {
        int t = t0 + i;
        const float2* p = (const float2*)(kv + (size_t)(b * TT + t) * NKV + h * 128);
        float2 kv0 = p[lane];
        float2 kv1 = p[lane + 32];
        float dot = wave_allreduce_add(q0 * kv0.x + q1 * kv1.x);
        float mn = fmaxf(m, dot);
        float e0 = __expf(m - mn), e1 = __expf(dot - mn);
        u  = u  * e0 + e1;
        w0 = w0 * e0 + kv0.y * e1;
        w1 = w1 * e0 + kv1.y * e1;
        m = mn;
        float inv = 1.0f / u;
        float* ho = hbuf + ((size_t)(h * BB + b) * TT + t) * DD;
        ho[lane]      = w0 * inv;
        ho[lane + 32] = w1 * inv;
    }
}

// ---------------------------------------------------------------------------
// Mean over heads + bf16 convert:  hmean[bt*64+d] = (1/16) sum_h hbuf[...]
// ---------------------------------------------------------------------------
__global__ void head_mean_kernel(const float* __restrict__ hbuf,
                                 unsigned short* __restrict__ hmean) {
    size_t idx = (size_t)blockIdx.x * blockDim.x + threadIdx.x; // bt*64 + d
    const size_t hstride = (size_t)BB * TT * DD;
    float s = 0.0f;
#pragma unroll
    for (int h = 0; h < HH; ++h) s += hbuf[h * hstride + idx];
    hmean[idx] = f32_to_bf16_rne(s * (1.0f / (float)HH));
}

// ---------------------------------------------------------------------------
// kernel_launch
// ---------------------------------------------------------------------------
extern "C" void kernel_launch(void* const* d_in, const int* in_sizes, int n_in,
                              void* d_out, int out_size, void* d_ws, size_t ws_size,
                              hipStream_t stream) {
    const float* inputs    = (const float*)d_in[0]; // (4,2048,1024)
    const float* kv_kernel = (const float*)d_in[1]; // (1024,16,64,2) == (1024,2048)
    const float* q_kernel  = (const float*)d_in[2]; // (16,64)
    const float* out_w     = (const float*)d_in[3]; // (1024,64)  -- already (N,K)
    const float* out_b     = (const float*)d_in[4]; // (1024)
    float* out = (float*)d_out;

    char* ws = (char*)d_ws;
    // Workspace layout (all 256B-aligned); total ~118 MB (fits L2-era ws).
    unsigned short* A_bf    = (unsigned short*)(ws + 0);          // 16 MB
    unsigned short* Wt_bf   = (unsigned short*)(ws + 16777216);   //  4 MB
    float*          kv_buf  = (float*)        (ws + 20971520);    // 64 MB
    float*          carry   = (float*)        (ws + 88080384);    // 272 KB
    float*          prefix  = (float*)        (ws + 88358912);    // 272 KB
    float*          hbuf    = (float*)        (ws + 88637440);    // 32 MB
    unsigned short* hm_bf   = (unsigned short*)(ws + 122191872);  //  1 MB
    unsigned short* ow_bf   = (unsigned short*)(ws + 123240448);  // 128 KB

    // 1) Convert activations to bf16
    {
        int n = MROWS * DIM_IN; // 8388608
        cvt_bf16_kernel<<<(n + 255) / 256, 256, 0, stream>>>(inputs, A_bf, n);
    }
    // 2) Transpose+convert kv weights to (N=2048, K=1024) bf16
    transpose_cvt_kernel<<<dim3(NKV / 32, DIM_IN / 32), dim3(32, 8), 0, stream>>>(kv_kernel, Wt_bf);
    // 3) Convert out_w (already (N,K) row-major) to bf16
    {
        int n = DIM_OUT * DD; // 65536
        cvt_bf16_kernel<<<(n + 255) / 256, 256, 0, stream>>>(out_w, ow_bf, n);
    }
    // 4) kv = inputs @ kv_kernel  (8192x1024 * 1024x2048) via WMMA bf16
    wmma_gemm_kernel<<<dim3(MROWS / 64, NKV / 128), 256, 0, stream>>>(
        A_bf, Wt_bf, kv_buf, nullptr, NKV, DIM_IN);
    // 5-7) Chunked associative scan (carries -> carry prefix -> apply)
    scan_carry_kernel<<<(BB * HH * NCHUNK) / 4, 128, 0, stream>>>(kv_buf, q_kernel, carry);
    scan_prefix_kernel<<<BB * HH, 32, 0, stream>>>(carry, prefix);
    scan_apply_kernel<<<(BB * HH * NCHUNK) / 4, 128, 0, stream>>>(kv_buf, q_kernel, prefix, hbuf);
    // 8) Mean over heads -> bf16
    head_mean_kernel<<<(MROWS * DD) / 256, 256, 0, stream>>>(hbuf, hm_bf);
    // 9) out = hmean @ out_w^T + out_b  (8192x64 * 64x1024) via WMMA bf16
    wmma_gemm_kernel<<<dim3(MROWS / 64, DIM_OUT / 128), 256, 0, stream>>>(
        hm_bf, ow_bf, out, out_b, DIM_OUT, DD);
}
